// Mamba2Wrapper_49426483642833
// MI455X (gfx1250) — matrix operand order repaired and checked by reading, not compiled, not run on hardware
//
#include <hip/hip_runtime.h>
#include <hip/hip_bf16.h>
#include <math.h>

// ---------------- problem constants (from reference) ----------------
#define D_MODEL   1024
#define D_STATE   128
#define D_CONV    4
#define HEADDIM   64
#define NHEADS    32
#define D_INNER   2048                  // EXPAND * D_MODEL
#define CONV_CH   2304                  // D_INNER + 2*D_STATE
#define D_IN_PROJ 4384                  // 2*D_INNER + 2*D_STATE + NHEADS
#define ZX_LD     4416                  // D_IN_PROJ padded to multiple of 64
#define BATCH     4
#define SEQLEN    2048
#define BL        (BATCH * SEQLEN)      // 8192
#define RMS_EPS   1e-5f

typedef __attribute__((ext_vector_type(16))) __bf16 v16bf;
typedef __attribute__((ext_vector_type(8)))  __bf16 v8bf;
typedef __attribute__((ext_vector_type(8)))  float  v8f;

// ---------------- f32 -> bf16 packer ----------------
__global__ void cvt_f32_bf16(const float* __restrict__ in,
                             __bf16* __restrict__ out, int n) {
    int i = blockIdx.x * blockDim.x + threadIdx.x;
    int stride = gridDim.x * blockDim.x;
    for (; i < n; i += stride) out[i] = (__bf16)in[i];
}

// Build a 16-half fragment from two 16-byte chunks.
__device__ __forceinline__ v16bf mk16(const __bf16* p0, const __bf16* p1) {
    v8bf lo = *(const v8bf*)p0;
    v8bf hi = *(const v8bf*)p1;
    v16bf r;
#pragma unroll
    for (int i = 0; i < 8; ++i) { r[i] = lo[i]; r[i + 8] = hi[i]; }
    return r;
}

// ---------------- WMMA bf16 GEMM:  C[M,N] = A[M,K] * B[N,K]^T ----------------
// Requirements: M % 32 == 0, N % 64 == 0, K % 64 == 0 (caller pads).
// One wave -> 32x64 output strip (2 M-tiles x 4 N-tiles, 8 f32 accumulators).
// Per CDNA5 ISA 16-bit layouts (CDNA5 ISA §7.12.2):
//   A 16x32 : lane l -> m = l&15 ; halves K = kb..kb+7 and kb+16..kb+23, kb=(l>>4)*8
//   B 32x16 : lane l -> n = l&15 ; halves K = kb2..kb2+7 and kb2+16..kb2+23, kb2=(l>>4)*8
//   C 16x16 : vgpr r, lane l -> m = r + 8*(l>>4), n = l&15
__global__ void __launch_bounds__(256)
wmma_gemm_bf16_nt(const __bf16* __restrict__ A, const __bf16* __restrict__ Bw,
                  float* __restrict__ C, int M, int N, int K) {
    const int lane = threadIdx.x & 31;
    // Force wave id scalar so all tile indexing is provably wave-uniform:
    // lane 0 of wave w has threadIdx.x == 32*w.
    const int wave = __builtin_amdgcn_readfirstlane((int)threadIdx.x) >> 5;
    const int ntg  = N >> 6;                      // 64-wide column groups
    const int gid  = blockIdx.x * 8 + wave;
    const int total = (M >> 5) * ntg;
    if (gid >= total) return;                     // scalar branch
    const int tm = gid / ntg;                     // 32-row strip index
    const int tn = gid % ntg;                     // 64-col strip index

    const int akb = (lane >> 4) << 3;             // 0 or 8
    const int bkb = (lane >> 4) << 3;             // 0 or 8
    const int lm  = lane & 15;

    const __bf16* arow0 = A + (size_t)((tm << 5) + lm) * K + akb;
    const __bf16* arow1 = arow0 + (size_t)16 * K;
    const __bf16* brow0 = Bw + (size_t)((tn << 6) + lm) * K + bkb;
    const __bf16* brow1 = brow0 + (size_t)16 * K;
    const __bf16* brow2 = brow0 + (size_t)32 * K;
    const __bf16* brow3 = brow0 + (size_t)48 * K;

    const v8f vzero = {0.f, 0.f, 0.f, 0.f, 0.f, 0.f, 0.f, 0.f};
    v8f acc[2][4];
#pragma unroll
    for (int mi = 0; mi < 2; ++mi)
#pragma unroll
        for (int j = 0; j < 4; ++j) acc[mi][j] = vzero;

    // ---- software-pipelined K loop, step 64, two named fragment sets ----
    v16bf a0_0 = mk16(arow0, arow0 + 16);
    v16bf a1_0 = mk16(arow1, arow1 + 16);
    v16bf b0_0 = mk16(brow0, brow0 + 16);
    v16bf b1_0 = mk16(brow1, brow1 + 16);
    v16bf b2_0 = mk16(brow2, brow2 + 16);
    v16bf b3_0 = mk16(brow3, brow3 + 16);

    for (int k0 = 0; k0 < K; k0 += 64) {
        __builtin_prefetch(arow0 + k0 + 256, 0, 3);
        __builtin_prefetch(arow1 + k0 + 256, 0, 3);
        // load fragment set B (k0+32) while WMMAs for set A issue
        v16bf a0_1 = mk16(arow0 + k0 + 32, arow0 + k0 + 48);
        v16bf a1_1 = mk16(arow1 + k0 + 32, arow1 + k0 + 48);
        v16bf b0_1 = mk16(brow0 + k0 + 32, brow0 + k0 + 48);
        v16bf b1_1 = mk16(brow1 + k0 + 32, brow1 + k0 + 48);
        v16bf b2_1 = mk16(brow2 + k0 + 32, brow2 + k0 + 48);
        v16bf b3_1 = mk16(brow3 + k0 + 32, brow3 + k0 + 48);

        acc[0][0] = __builtin_amdgcn_wmma_f32_16x16x32_bf16(false, a0_0, false, b0_0, (short)0, acc[0][0], false, false);
        acc[0][1] = __builtin_amdgcn_wmma_f32_16x16x32_bf16(false, a0_0, false, b1_0, (short)0, acc[0][1], false, false);
        acc[0][2] = __builtin_amdgcn_wmma_f32_16x16x32_bf16(false, a0_0, false, b2_0, (short)0, acc[0][2], false, false);
        acc[0][3] = __builtin_amdgcn_wmma_f32_16x16x32_bf16(false, a0_0, false, b3_0, (short)0, acc[0][3], false, false);
        acc[1][0] = __builtin_amdgcn_wmma_f32_16x16x32_bf16(false, a1_0, false, b0_0, (short)0, acc[1][0], false, false);
        acc[1][1] = __builtin_amdgcn_wmma_f32_16x16x32_bf16(false, a1_0, false, b1_0, (short)0, acc[1][1], false, false);
        acc[1][2] = __builtin_amdgcn_wmma_f32_16x16x32_bf16(false, a1_0, false, b2_0, (short)0, acc[1][2], false, false);
        acc[1][3] = __builtin_amdgcn_wmma_f32_16x16x32_bf16(false, a1_0, false, b3_0, (short)0, acc[1][3], false, false);

        if (k0 + 64 < K) {
            // load next set A (k0+64) while WMMAs for set B issue
            a0_0 = mk16(arow0 + k0 + 64, arow0 + k0 + 80);
            a1_0 = mk16(arow1 + k0 + 64, arow1 + k0 + 80);
            b0_0 = mk16(brow0 + k0 + 64, brow0 + k0 + 80);
            b1_0 = mk16(brow1 + k0 + 64, brow1 + k0 + 80);
            b2_0 = mk16(brow2 + k0 + 64, brow2 + k0 + 80);
            b3_0 = mk16(brow3 + k0 + 64, brow3 + k0 + 80);
        }

        acc[0][0] = __builtin_amdgcn_wmma_f32_16x16x32_bf16(false, a0_1, false, b0_1, (short)0, acc[0][0], false, false);
        acc[0][1] = __builtin_amdgcn_wmma_f32_16x16x32_bf16(false, a0_1, false, b1_1, (short)0, acc[0][1], false, false);
        acc[0][2] = __builtin_amdgcn_wmma_f32_16x16x32_bf16(false, a0_1, false, b2_1, (short)0, acc[0][2], false, false);
        acc[0][3] = __builtin_amdgcn_wmma_f32_16x16x32_bf16(false, a0_1, false, b3_1, (short)0, acc[0][3], false, false);
        acc[1][0] = __builtin_amdgcn_wmma_f32_16x16x32_bf16(false, a1_1, false, b0_1, (short)0, acc[1][0], false, false);
        acc[1][1] = __builtin_amdgcn_wmma_f32_16x16x32_bf16(false, a1_1, false, b1_1, (short)0, acc[1][1], false, false);
        acc[1][2] = __builtin_amdgcn_wmma_f32_16x16x32_bf16(false, a1_1, false, b2_1, (short)0, acc[1][2], false, false);
        acc[1][3] = __builtin_amdgcn_wmma_f32_16x16x32_bf16(false, a1_1, false, b3_1, (short)0, acc[1][3], false, false);
    }

    // ---- store: rows tm*32 + mi*16 + r + 8*(lane>>4), col tn*64 + j*16 + lm ----
    const int rhalf = (lane >> 4) << 3;
#pragma unroll
    for (int mi = 0; mi < 2; ++mi) {
        const int crow0 = (tm << 5) + (mi << 4) + rhalf;
#pragma unroll
        for (int j = 0; j < 4; ++j) {
            const int ccol = (tn << 6) + (j << 4) + lm;
#pragma unroll
            for (int r = 0; r < 8; ++r)
                C[(size_t)(crow0 + r) * N + ccol] = acc[mi][j][r];
        }
    }
}

// ---------------- dt = softplus(raw + dt_bias) ----------------
__global__ void dt_softplus(const float* __restrict__ zx,
                            const float* __restrict__ dt_bias,
                            float* __restrict__ dtb) {
    int i = blockIdx.x * blockDim.x + threadIdx.x;      // BL*NHEADS
    if (i >= BL * NHEADS) return;
    const int row = i >> 5, h = i & 31;
    float x = zx[(size_t)row * ZX_LD + (D_INNER + CONV_CH) + h] + dt_bias[h];
    dtb[i] = (x > 20.f) ? x : log1pf(expf(x));
}

// ---------------- depthwise causal conv (k=4) + bias + SiLU ----------------
__global__ void conv_silu(const float* __restrict__ zx,
                          const float* __restrict__ cw,
                          const float* __restrict__ cb,
                          float* __restrict__ out) {
    long long i = (long long)blockIdx.x * blockDim.x + threadIdx.x;  // BL*CONV_CH
    if (i >= (long long)BL * CONV_CH) return;
    const int ch  = (int)(i % CONV_CH);
    const int row = (int)(i / CONV_CH);                 // b*L + l
    const int l   = row & (SEQLEN - 1);
    float acc = cb[ch];
#pragma unroll
    for (int j = 0; j < D_CONV; ++j) {
        const int ls = l - (D_CONV - 1) + j;
        if (ls >= 0)
            acc += cw[ch * D_CONV + j] *
                   zx[(size_t)(row - (D_CONV - 1) + j) * ZX_LD + D_INNER + ch];
    }
    out[i] = acc / (1.f + expf(-acc));                  // SiLU
}

// ---------------- sequential SSM scan: one block per (b, head) ----------------
// h[64,128] kept in registers: 256 threads x 32 f32. Thread t -> p = t>>2,
// n-slice = (t&3)*32 .. +32. y reduced over the 4 n-slices via wave32 shfl_xor.
__global__ void __launch_bounds__(256)
ssm_scan(const float* __restrict__ xbc, const float* __restrict__ dtb,
         const float* __restrict__ A_log, const float* __restrict__ Dv,
         float* __restrict__ ybuf) {
    const int bh = blockIdx.x;                          // 0..127
    const int b  = bh >> 5;
    const int h  = bh & 31;
    const int t  = threadIdx.x;
    const int p  = t >> 2;
    const int n0 = (t & 3) << 5;

    __shared__ float xs_s[HEADDIM];
    __shared__ float B_s[D_STATE];
    __shared__ float C_s[D_STATE];
    __shared__ float sc_s[2];

    const float Ah = -expf(A_log[h]);
    const float Dh = Dv[h];

    float hreg[32];
#pragma unroll
    for (int j = 0; j < 32; ++j) hreg[j] = 0.f;

    const float* base = xbc + (size_t)b * SEQLEN * CONV_CH;

    for (int l = 0; l < SEQLEN; ++l) {
        const float* r = base + (size_t)l * CONV_CH;
        // stage per-step operands in LDS (320 loads across 256 threads)
        if (t < HEADDIM) {
            xs_s[t] = r[h * HEADDIM + t];
        } else if (t < HEADDIM + D_STATE) {
            B_s[t - HEADDIM] = r[D_INNER + (t - HEADDIM)];
        } else {
            const int k = (t - (HEADDIM + D_STATE)) << 1;
            C_s[k]     = r[D_INNER + D_STATE + k];
            C_s[k + 1] = r[D_INNER + D_STATE + k + 1];
        }
        if (t == 0) {
            const float dtv = dtb[(size_t)(b * SEQLEN + l) * NHEADS + h];
            sc_s[0] = dtv;
            sc_s[1] = expf(dtv * Ah);
        }
        // prefetch next step's operands while this step computes
        if (l + 1 < SEQLEN) {
            const float* rn = r + CONV_CH;
            if (t < HEADDIM)                       __builtin_prefetch(rn + h * HEADDIM + t, 0, 3);
            else if (t < HEADDIM + D_STATE)        __builtin_prefetch(rn + D_INNER + (t - HEADDIM), 0, 3);
            else __builtin_prefetch(rn + D_INNER + D_STATE + ((t - (HEADDIM + D_STATE)) << 1), 0, 3);
        }
        __syncthreads();

        const float a  = sc_s[1];
        const float sx = sc_s[0] * xs_s[p];
        float acc = 0.f;
#pragma unroll
        for (int j = 0; j < 32; ++j) {
            hreg[j] = fmaf(hreg[j], a, sx * B_s[n0 + j]);
            acc     = fmaf(hreg[j], C_s[n0 + j], acc);
        }
        acc += __shfl_xor(acc, 1);                      // wave32 cross-lane reduce
        acc += __shfl_xor(acc, 2);
        if ((t & 3) == 0)
            ybuf[(size_t)(b * SEQLEN + l) * D_INNER + h * HEADDIM + p] =
                acc + Dh * xs_s[p];
        __syncthreads();
    }
}

// ---------------- y * silu(z), RMSNorm(2048), emit bf16 ----------------
__global__ void __launch_bounds__(256)
gate_norm(const float* __restrict__ ybuf, const float* __restrict__ zx,
          const float* __restrict__ nw, __bf16* __restrict__ out) {
    const int row = blockIdx.x;
    const int t   = threadIdx.x;
    __shared__ float red[256];
    float v[8];
    float ss = 0.f;
#pragma unroll
    for (int i = 0; i < 8; ++i) {
        const int c = t + i * 256;
        const float y = ybuf[(size_t)row * D_INNER + c];
        const float z = zx[(size_t)row * ZX_LD + c];
        const float g = z / (1.f + expf(-z));           // silu(z)
        const float val = y * g;
        v[i] = val;
        ss += val * val;
    }
    red[t] = ss;
    __syncthreads();
    for (int s = 128; s > 0; s >>= 1) {
        if (t < s) red[t] += red[t + s];
        __syncthreads();
    }
    const float scale = rsqrtf(red[0] / (float)D_INNER + RMS_EPS);
#pragma unroll
    for (int i = 0; i < 8; ++i) {
        const int c = t + i * 256;
        out[(size_t)row * D_INNER + c] = (__bf16)(v[i] * scale * nw[c]);
    }
}

// ---------------- host orchestration ----------------
extern "C" void kernel_launch(void* const* d_in, const int* in_sizes, int n_in,
                              void* d_out, int out_size, void* d_ws, size_t ws_size,
                              hipStream_t stream) {
    const float* x       = (const float*)d_in[0];   // [BL, 1024]
    const float* W_in    = (const float*)d_in[1];   // [4384, 1024]
    const float* conv_w  = (const float*)d_in[2];   // [2304, 4]
    const float* conv_b  = (const float*)d_in[3];   // [2304]
    const float* dt_bias = (const float*)d_in[4];   // [32]
    const float* A_log   = (const float*)d_in[5];   // [32]
    const float* Dv      = (const float*)d_in[6];   // [32]
    const float* norm_w  = (const float*)d_in[7];   // [2048]
    const float* W_out   = (const float*)d_in[8];   // [1024, 2048]
    float* out = (float*)d_out;                     // [BL, 1024] f32

    char* ws = (char*)d_ws;
    size_t off = 0;
    auto carve = [&](size_t bytes) -> void* {
        void* p = ws + off;
        off = (off + bytes + 255) & ~(size_t)255;
        return p;
    };
    __bf16* xb    = (__bf16*)carve((size_t)BL * D_MODEL * 2);          // 16.8 MB
    __bf16* Winb  = (__bf16*)carve((size_t)ZX_LD * D_MODEL * 2);       //  9.0 MB (padded rows)
    __bf16* Woutb = (__bf16*)carve((size_t)D_MODEL * D_INNER * 2);     //  4.2 MB
    float*  zx    = (float*) carve((size_t)BL * ZX_LD * 4);            // 144.7 MB (padded cols)
    float*  dtb   = (float*) carve((size_t)BL * NHEADS * 4);           //  1.0 MB
    float*  xbcc  = (float*) carve((size_t)BL * CONV_CH * 4);          // 75.5 MB
    float*  ybuf  = (float*) carve((size_t)BL * D_INNER * 4);          // 67.1 MB
    __bf16* yb    = (__bf16*)carve((size_t)BL * D_INNER * 2);          // 33.6 MB

    // 1) bf16 packing
    {
        int n = BL * D_MODEL;
        cvt_f32_bf16<<<(n + 255) / 256, 256, 0, stream>>>(x, xb, n);
        n = D_IN_PROJ * D_MODEL;                 // pad rows of Winb stay unwritten; they only feed unused zx pad cols
        cvt_f32_bf16<<<(n + 255) / 256, 256, 0, stream>>>(W_in, Winb, n);
        n = D_MODEL * D_INNER;
        cvt_f32_bf16<<<(n + 255) / 256, 256, 0, stream>>>(W_out, Woutb, n);
    }

    // 2) in_proj: zx[BL,4416] = xb[BL,1024] * Winb[4416,1024]^T  (cols >=4384 are pad)
    {
        const int waves = (BL / 32) * (ZX_LD / 64);         // 256 * 69 = 17664
        wmma_gemm_bf16_nt<<<(waves + 7) / 8, 256, 0, stream>>>(
            xb, Winb, zx, BL, ZX_LD, D_MODEL);
    }

    // 3) dt softplus
    {
        const int n = BL * NHEADS;
        dt_softplus<<<(n + 255) / 256, 256, 0, stream>>>(zx, dt_bias, dtb);
    }

    // 4) depthwise conv + SiLU
    {
        const long long n = (long long)BL * CONV_CH;
        conv_silu<<<(unsigned)((n + 255) / 256), 256, 0, stream>>>(
            zx, conv_w, conv_b, xbcc);
    }

    // 5) sequential SSM scan
    ssm_scan<<<BATCH * NHEADS, 256, 0, stream>>>(xbcc, dtb, A_log, Dv, ybuf);

    // 6) gate + RMSNorm -> bf16
    gate_norm<<<BL, 256, 0, stream>>>(ybuf, zx, norm_w, yb);

    // 7) out_proj: out[BL,1024] = yb[BL,2048] * Woutb[1024,2048]^T
    {
        const int waves = (BL / 32) * (D_MODEL / 64);       // 256 * 16 = 4096
        wmma_gemm_bf16_nt<<<(waves + 7) / 8, 256, 0, stream>>>(
            yb, Woutb, out, BL, D_MODEL, D_INNER);
    }
}
